// GNNFeatureExtractor_56006373540168
// MI455X (gfx1250) — matrix operand order repaired and checked by reading, not compiled, not run on hardware
//
#include <hip/hip_runtime.h>
#include <hip/hip_bf16.h>

// ---------------------------------------------------------------------------
// GNN (2x dense GAT over 4800 fully-connected nodes) + MLP head, fused
// flash-attention style for MI455X (gfx1250, wave32, WMMA f16 16x16x32).
// K-loop split over 5 waves/block for latency hiding; LDS partial reduction.
// ---------------------------------------------------------------------------

#define NN      4800      // B*JOBS nodes
#define FEAT    15
#define NH      3
#define DD1     64
#define DD2     128
#define C1      192       // NH*DD1
#define C2      384       // NH*DD2
#define BATCH   32
#define JOBS    150
#define HID     64

typedef _Float16 v16h __attribute__((ext_vector_type(16)));
typedef _Float16 v8h  __attribute__((ext_vector_type(8)));
typedef float    v8f  __attribute__((ext_vector_type(8)));

// ---- layer-1 projection: h1 = flat @ W1 ; scores s1,t1 ; f16 transposed copy
__global__ void k_prep1(const float* __restrict__ obs, const float* __restrict__ W1,
                        const float* __restrict__ asrc, const float* __restrict__ adst,
                        _Float16* __restrict__ hT1, float* __restrict__ s1,
                        float* __restrict__ t1) {
  __shared__ float sh[C1];
  int node = blockIdx.x, c = threadIdx.x;              // 192 threads
  const float* x = obs + node * FEAT;
  float acc = 0.f;
#pragma unroll
  for (int f = 0; f < FEAT; ++f) acc += x[f] * W1[f * C1 + c];
  sh[c] = acc;
  hT1[(size_t)c * NN + node] = (_Float16)acc;          // [h*64+d][node]
  __syncthreads();
  if (c < 2 * NH) {
    int head = c % NH;  bool isT = c >= NH;
    const float* a = isT ? adst : asrc;
    float dot = 0.f;
    for (int d = 0; d < DD1; ++d) dot += sh[head * DD1 + d] * a[head * DD1 + d];
    (isT ? t1 : s1)[node * NH + head] = dot;
  }
}

// ---- per-head global max of t (one block per head)
__global__ void k_colmax(const float* __restrict__ t, float* __restrict__ tmax) {
  __shared__ float red[256];
  int h = blockIdx.x, tid = threadIdx.x;
  float m = -1e30f;
  for (int j = tid; j < NN; j += 256) m = fmaxf(m, t[j * NH + h]);
  red[tid] = m; __syncthreads();
  for (int s = 128; s > 0; s >>= 1) {
    if (tid < s) red[tid] = fmaxf(red[tid], red[tid + s]);
    __syncthreads();
  }
  if (tid == 0) tmax[h] = red[0];
}

// ---- softmax factor precompute: exp(lrelu(s+t)-m) = pos ? A_i*C_j : B_i*D_j
__global__ void k_coef(const float* __restrict__ s, const float* __restrict__ t,
                       const float* __restrict__ tmax,
                       float* __restrict__ A, float* __restrict__ B,
                       float* __restrict__ C, float* __restrict__ D) {
  int i = blockIdx.x * blockDim.x + threadIdx.x;
  if (i >= NN * NH) return;
  int h = i % NH;
  float tm = tmax[h], sv = s[i], tv = t[i];
  float x = sv + tm;
  float m = x > 0.f ? x : 0.2f * x;                    // row max of lrelu logits
  A[i] = __expf(sv + tm - m);
  B[i] = __expf(0.2f * sv - m);
  C[i] = __expf(tv - tm);
  D[i] = __expf(0.2f * tv);
}

// ---- fused attention aggregation: out[i,h,:] = softmax_j(lrelu(s_i+t_j)) @ h[j,h,:]
// 5 waves per (16-row tile, head); each wave owns disjoint 32-wide j-chunks,
// A-tile built in registers from factorized weights, B streamed f16, WMMA f32
// accumulate; per-wave partials + Z reduced through LDS, rcp-normalized.
template <int DHEAD, bool ELU, int HSTRIDE, int RSTRIDE>
__global__ void __launch_bounds__(160) k_agg(
    const _Float16* __restrict__ hT, const float* __restrict__ s,
    const float* __restrict__ t, const float* __restrict__ Ac,
    const float* __restrict__ Bc, const float* __restrict__ Cc,
    const float* __restrict__ Dc, float* __restrict__ out) {
  constexpr int NT = DHEAD / 16;
  constexpr int NW = 5;                                 // waves per block
  constexpr int ITER = (NN / 32) / NW;                  // 30 chunks per wave
  int it = blockIdx.x, h = blockIdx.y;
  int tid = threadIdx.x;
  int wv = tid >> 5, l = tid & 31;
  int n15 = l & 15;                                     // A row / B col / C col
  int kbA = (l >= 16) ? 8 : 0;                          // A K-group base
  int kbB = (l >= 16) ? 8 : 0;                          // B K-group base
  int i0 = it * 16;
  int ni = (i0 + n15) * NH + h;
  float sI = s[ni], AI = Ac[ni], BI = Bc[ni];

  __shared__ float Tt[NW][32], Ct[NW][32], Dt[NW][32];
  __shared__ float part[NW][NT][8][32];
  __shared__ float zl[NW][32];
  __shared__ float rZ[16];

  v8f acc[NT] = {};
  float zacc = 0.f;

  for (int k = 0; k < ITER; ++k) {
    int j0 = (wv + k * NW) * 32;
    int j = (j0 + l) * NH + h;
    Tt[wv][l] = t[j];  Ct[wv][l] = Cc[j];  Dt[wv][l] = Dc[j];
    __syncthreads();

    // prefetch next chunk's value rows toward this WGP (global_prefetch_b8)
    if (k + 1 < ITER) {
      const _Float16* pf =
          hT + (size_t)(h * DHEAD + n15) * NN + (j0 + NW * 32) + kbB;
      __builtin_prefetch(pf, 0, 0);
    }

    // A fragment: 16x32 f16 per ISA layout (lane row = l&15, K in 2 runs of 8)
    v16h a;
#pragma unroll
    for (int e = 0; e < 16; ++e) {
      int K = (e < 8 ? e : e + 8) + kbA;
      float tv = Tt[wv][K];
      bool pos = (sI + tv) > 0.f;
      float wgt = (pos ? AI : BI) * (pos ? Ct[wv][K] : Dt[wv][K]);
      zacc += wgt;
      a[e] = (_Float16)wgt;
    }

#pragma unroll
    for (int q = 0; q < NT; ++q) {
      const _Float16* bp =
          hT + (size_t)(h * DHEAD + q * 16 + n15) * NN + j0 + kbB;
      v8h blo = *(const v8h*)bp;
      v8h bhi = *(const v8h*)(bp + 16);
      v16h b;
#pragma unroll
      for (int e = 0; e < 8; ++e) { b[e] = blo[e]; b[e + 8] = bhi[e]; }
      acc[q] = __builtin_amdgcn_wmma_f32_16x16x32_f16(
          false, a, false, b, (short)0, acc[q], false, false);
    }
  }

  // stage per-wave partials (accumulators + Z) to LDS
  zl[wv][l] = zacc;
#pragma unroll
  for (int q = 0; q < NT; ++q)
#pragma unroll
    for (int r = 0; r < 8; ++r) part[wv][q][r][l] = acc[q][r];
  __syncthreads();

  // Z per output row m lives in lanes m and m+16 of every wave
  if (tid < 16) {
    float zz = 0.f;
#pragma unroll
    for (int w2 = 0; w2 < NW; ++w2) zz += zl[w2][tid] + zl[w2][tid + 16];
    rZ[tid] = __builtin_amdgcn_rcpf(zz);
  }
  __syncthreads();

  // reduce partials, normalize, activate, store
  for (int idx = tid; idx < NT * 256; idx += NW * 32) {
    int q = idx >> 8, rem = idx & 255;
    int r = rem >> 5, lane = rem & 31;
    int m = r + ((lane >= 16) ? 8 : 0);                 // C/D VGPR layout
    int n = lane & 15;
    float v = 0.f;
#pragma unroll
    for (int w2 = 0; w2 < NW; ++w2) v += part[w2][q][r][lane];
    v *= rZ[m];
    if (ELU) v = v > 0.f ? v : (__expf(v) - 1.f);
    out[(size_t)h * HSTRIDE + (size_t)(i0 + m) * RSTRIDE + q * 16 + n] = v;
  }
}

// ---- layer-2 projection: h2 = act1 @ W2 ; scores ; f16 transposed copy
__global__ void k_h2(const float* __restrict__ act1, const float* __restrict__ W2,
                     const float* __restrict__ asrc, const float* __restrict__ adst,
                     _Float16* __restrict__ hT2, float* __restrict__ s2,
                     float* __restrict__ t2) {
  __shared__ float xin[C1];
  __shared__ float sh[C2];
  int node = blockIdx.x, c = threadIdx.x;               // 384 threads
  if (c < C1) xin[c] = act1[node * C1 + c];
  __syncthreads();
  float acc = 0.f;
  for (int k = 0; k < C1; ++k) acc += xin[k] * W2[k * C2 + c];
  sh[c] = acc;
  hT2[(size_t)c * NN + node] = (_Float16)acc;
  __syncthreads();
  if (c < 2 * NH) {
    int head = c % NH;  bool isT = c >= NH;
    const float* a = isT ? adst : asrc;
    float dot = 0.f;
    for (int d = 0; d < DD2; ++d) dot += sh[head * DD2 + d] * a[head * DD2 + d];
    (isT ? t2 : s2)[node * NH + head] = dot;
  }
}

// ---- head-mean + concat raw feats + 2-layer relu MLP
__global__ void k_mlp(const float* __restrict__ out2, const float* __restrict__ obs,
                      const float* __restrict__ P1, const float* __restrict__ b1,
                      const float* __restrict__ P2, const float* __restrict__ b2,
                      float* __restrict__ f2buf) {
  __shared__ float g[DD2 + FEAT];
  __shared__ float f1[2 * HID];
  int node = blockIdx.x, c = threadIdx.x;               // 128 threads
  const size_t HS = (size_t)NN * DD2;
  g[c] = (out2[node * DD2 + c] + out2[HS + node * DD2 + c] +
          out2[2 * HS + node * DD2 + c]) * (1.f / 3.f);
  if (c < FEAT) g[DD2 + c] = obs[node * FEAT + c];
  __syncthreads();
  float acc = b1[c];
  for (int k = 0; k < DD2 + FEAT; ++k) acc += g[k] * P1[k * (2 * HID) + c];
  f1[c] = fmaxf(acc, 0.f);
  __syncthreads();
  if (c < HID) {
    float a2 = b2[c];
    for (int k = 0; k < 2 * HID; ++k) a2 += f1[k] * P2[k * HID + c];
    f2buf[node * HID + c] = fmaxf(a2, 0.f);
  }
}

// ---- mean-pool over jobs -> features [32,64]
__global__ void k_pool(const float* __restrict__ f2buf, float* __restrict__ out) {
  int b = blockIdx.x, d = threadIdx.x;                  // 64 threads
  float s = 0.f;
  for (int j = 0; j < JOBS; ++j) s += f2buf[(b * JOBS + j) * HID + d];
  out[b * HID + d] = s * (1.f / (float)JOBS);
}

__global__ void k_mask(const float* __restrict__ mask, float* __restrict__ out) {
  int i = blockIdx.x * blockDim.x + threadIdx.x;
  if (i < BATCH * (JOBS + 1)) out[BATCH * HID + i] = mask[i];
}

// ---------------------------------------------------------------------------
extern "C" void kernel_launch(void* const* d_in, const int* in_sizes, int n_in,
                              void* d_out, int out_size, void* d_ws, size_t ws_size,
                              hipStream_t stream) {
  const float* obs   = (const float*)d_in[0];
  const float* amask = (const float*)d_in[1];
  const float* W1    = (const float*)d_in[2];
  const float* as1   = (const float*)d_in[3];
  const float* ad1   = (const float*)d_in[4];
  const float* W2    = (const float*)d_in[5];
  const float* as2   = (const float*)d_in[6];
  const float* ad2   = (const float*)d_in[7];
  const float* P1    = (const float*)d_in[8];
  const float* b1    = (const float*)d_in[9];
  const float* P2    = (const float*)d_in[10];
  const float* b2    = (const float*)d_in[11];
  float* out = (float*)d_out;

  char* w = (char*)d_ws;
  size_t off = 0;
  auto carve = [&](size_t bytes) -> void* {
    void* p = w + off;
    off = (off + bytes + 255) & ~(size_t)255;
    return p;
  };
  float*    act1  = (float*)carve((size_t)NN * C1 * 4);
  _Float16* hT1   = (_Float16*)carve((size_t)NH * DD1 * NN * 2);
  float* s1 = (float*)carve(NN * NH * 4);  float* t1 = (float*)carve(NN * NH * 4);
  float* A1 = (float*)carve(NN * NH * 4);  float* B1 = (float*)carve(NN * NH * 4);
  float* Cx1 = (float*)carve(NN * NH * 4); float* Dx1 = (float*)carve(NN * NH * 4);
  float* tm1 = (float*)carve(256);
  _Float16* hT2 = (_Float16*)carve((size_t)NH * DD2 * NN * 2);
  float* s2 = (float*)carve(NN * NH * 4);  float* t2 = (float*)carve(NN * NH * 4);
  float* A2 = (float*)carve(NN * NH * 4);  float* B2 = (float*)carve(NN * NH * 4);
  float* Cx2 = (float*)carve(NN * NH * 4); float* Dx2 = (float*)carve(NN * NH * 4);
  float* tm2 = (float*)carve(256);
  float* out2 = (float*)carve((size_t)NH * NN * DD2 * 4);
  float* f2b  = (float*)carve((size_t)NN * HID * 4);
  (void)ws_size; (void)in_sizes; (void)n_in; (void)out_size;

  // layer 1
  k_prep1<<<NN, C1, 0, stream>>>(obs, W1, as1, ad1, hT1, s1, t1);
  k_colmax<<<NH, 256, 0, stream>>>(t1, tm1);
  k_coef<<<(NN * NH + 255) / 256, 256, 0, stream>>>(s1, t1, tm1, A1, B1, Cx1, Dx1);
  k_agg<DD1, true, DD1, C1><<<dim3(NN / 16, NH), 160, 0, stream>>>(
      hT1, s1, t1, A1, B1, Cx1, Dx1, act1);

  // layer 2
  k_h2<<<NN, C2, 0, stream>>>(act1, W2, as2, ad2, hT2, s2, t2);
  k_colmax<<<NH, 256, 0, stream>>>(t2, tm2);
  k_coef<<<(NN * NH + 255) / 256, 256, 0, stream>>>(s2, t2, tm2, A2, B2, Cx2, Dx2);
  k_agg<DD2, false, NN * DD2, DD2><<<dim3(NN / 16, NH), 160, 0, stream>>>(
      hT2, s2, t2, A2, B2, Cx2, Dx2, out2);

  // head
  k_mlp<<<NN, 2 * HID, 0, stream>>>(out2, obs, P1, b1, P2, b2, f2b);
  k_pool<<<BATCH, HID, 0, stream>>>(f2b, out);
  k_mask<<<(BATCH * (JOBS + 1) + 255) / 256, 256, 0, stream>>>(amask, out);
}
